// STABR_song_encoder_26697516712556
// MI455X (gfx1250) — matrix operand order repaired and checked by reading, not compiled, not run on hardware
//
#include <hip/hip_runtime.h>

typedef __attribute__((ext_vector_type(16))) _Float16 v16h;
typedef __attribute__((ext_vector_type(8)))  _Float16 v8h;
typedef __attribute__((ext_vector_type(8)))  float    v8f;

#define B_SZ 1024
#define T_SZ 200
#define E_SZ 50
#define H_SZ 64
#define BT   (B_SZ * T_SZ)        // 204800 tokens, divisible by 128
#define G3H  192                  // 3*H
#define MT_ATT 13                 // ceil(200/16) time tiles for attention

// ---------------------------------------------------------------------------
// A-fragment (16-bit A 16x32, ISA 7.12.2):
//   lane = (row%16) + 16*lanehi ; koff in [0,32): lanehi=(koff>>3)&1,
//   m = (koff&7) + ((koff&16)?8:0)   (inverse: koff = m + (m&8) + 8*lanehi)
// B-fragment (32x16): lane = col(0..15) + 16*g ; k = 32*kb + 8*g + (m&7) + 16*(m>>3) (scan hB: 16*g + m)
// C layout: vgpr r: row = r + 8*(lane>=16), col = lane%16
// For an 8-aligned run of 8 consecutive k (or col) values the fragment slots
// are contiguous -> every hot access below is a 16B/32B vector op.
// ---------------------------------------------------------------------------

__global__ void k_gather(const int* __restrict__ ids, const float* __restrict__ emb,
                         _Float16* __restrict__ xfrag) {
  int tok = blockIdx.x * blockDim.x + threadIdx.x;
  if (tok >= BT) return;
  int id = ids[tok];
  int mtile = tok >> 4, rlo = tok & 15;
  const float* src = emb + (long)id * E_SZ;
  for (int k = 0; k < 64; ++k) {
    float v = (k < E_SZ) ? src[k] : 0.0f;
    int kb = k >> 5, koff = k & 31;
    int lane = rlo + (((koff >> 3) & 1) ? 16 : 0);
    int m = (koff & 7) + ((koff & 16) ? 8 : 0);
    xfrag[(((long)(mtile * 2 + kb)) * 32 + lane) * 16 + m] = (_Float16)v;
  }
}

// Pack f32 weight [K,N] row-major into f16 B-fragment order [nt][kb][lane][m].
__global__ void k_packB(const float* __restrict__ W, _Float16* __restrict__ dst,
                        int K, int N, int Kp) {
  int idx = blockIdx.x * blockDim.x + threadIdx.x;
  int kbn = Kp / 32;
  int total = (N / 16) * kbn * 512;
  if (idx >= total) return;
  int m = idx & 15, lane = (idx >> 4) & 31;
  int rest = idx >> 9;
  int kb = rest % kbn, nt = rest / kbn;
  int n = nt * 16 + (lane & 15);
  int k = kb * 32 + ((lane >= 16) ? 8 : 0) + (m & 7) + ((m & 8) ? 16 : 0);
  float v = (k < K) ? W[(long)k * N + n] : 0.0f;
  dst[idx] = (_Float16)v;
}

// Pack Wh^T [192 rows(out cols) x 64 k(h-dim)] into f16 A-fragment order
// [mt(12)][kb(2)][lane][m], reading Wh stored as [64,192] row-major.
__global__ void k_packA_T(const float* __restrict__ Wh, _Float16* __restrict__ dst) {
  int idx = blockIdx.x * blockDim.x + threadIdx.x;
  if (idx >= 12 * 2 * 512) return;
  int m = idx & 15, lane = (idx >> 4) & 31;
  int rest = idx >> 9;
  int kb = rest & 1, mt = rest >> 1;
  int row = mt * 16 + (lane & 15);                 // output column of Wh
  int koff = m + ((lane >= 16) ? 16 : 0);
  int k = kb * 32 + koff;                          // h-dim index
  dst[idx] = (_Float16)Wh[(long)k * G3H + row];
}

// gx = x @ Wx + bi  -> f16 [BT,192] linear. 256 thr (8 waves), wave = one M tile.
__global__ void __launch_bounds__(256) k_gemm_gx(const _Float16* __restrict__ xfrag,
                          const _Float16* __restrict__ wfrag,
                          const float* __restrict__ bi,
                          _Float16* __restrict__ gx) {
  int lane = threadIdx.x & 31, wave = threadIdx.x >> 5;
  int mtile = blockIdx.x * 8 + wave;                 // BT/16 = 12800 tiles
  v16h a0 = *(const v16h*)(xfrag + (((long)mtile * 2 + 0) * 32 + lane) * 16);
  v16h a1 = *(const v16h*)(xfrag + (((long)mtile * 2 + 1) * 32 + lane) * 16);
  for (int nt = 0; nt < 12; ++nt) {
    v16h b0 = *(const v16h*)(wfrag + (((nt * 2 + 0) * 32) + lane) * 16);
    v16h b1 = *(const v16h*)(wfrag + (((nt * 2 + 1) * 32) + lane) * 16);
    v8f c = {};
    c = __builtin_amdgcn_wmma_f32_16x16x32_f16(false, a0, false, b0, (short)0, c, false, false);
    c = __builtin_amdgcn_wmma_f32_16x16x32_f16(false, a1, false, b1, (short)0, c, false, false);
    int col = nt * 16 + (lane & 15);
    float bias = bi[col];
    for (int r = 0; r < 8; ++r) {
      int row = mtile * 16 + r + ((lane >= 16) ? 8 : 0);
      gx[(long)row * G3H + col] = (_Float16)(c[r] + bias);
    }
  }
}

// GRU scan, transposed recurrence: gh^T = Wh^T @ h^T.
// 64 blocks x 128 threads per direction; block owns 16 batch rows.
// Wave w computes M-tiles {w, w+4, w+8} (gates z,r,n for h-cols [16w,16w+16)),
// so lane (w,l) owns batch row l%16 and 8 consecutive h-cols, keeping all three
// gate values AND the f32 state in registers: no gh LDS round-trip, single
// barrier per step with ping-pong h^T B-fragment buffers. gx stream is register
// double-buffered and prefetched 8 steps ahead.
__global__ void __launch_bounds__(128) k_scan(const _Float16* __restrict__ gx,
                       const _Float16* __restrict__ whA,
                       const float* __restrict__ br,
                       const int* __restrict__ ids,
                       float* __restrict__ out32,
                       _Float16* __restrict__ outfrag,
                       float* __restrict__ hout,
                       int forward, int coloff) {
  __shared__ __align__(32) _Float16 hB[2][2 * 32 * 16];   // h^T B-frags, ping-pong
  int tid = threadIdx.x, lane = tid & 31, wave = tid >> 5;
  int brow = blockIdx.x * 16;
  for (int i = tid; i < 2 * 32 * 16; i += 128) hB[0][i] = (_Float16)0.0f;

  // Preload Wh^T A-fragments: wave w -> tiles {w, w+4, w+8}.
  v16h afr[3][2];
  for (int j = 0; j < 3; ++j)
    for (int kb = 0; kb < 2; ++kb)
      afr[j][kb] = *(const v16h*)(whA + ((((wave + 4 * j) * 2 + kb) * 32) + lane) * 16);

  int hi = (lane >= 16) ? 1 : 0;
  int bn = lane & 15;                    // batch row owned by this lane
  int cb = wave * 16 + hi * 8;           // h-col base (8-aligned, 8 consecutive)
  int b = brow + bn;
  const _Float16* gxrow = gx + (long)b * T_SZ * G3H;
  const int* idrow = ids + (long)b * T_SZ;
  // biases (constant per lane)
  v8f brz = *(const v8f*)(br + cb);
  v8f brr = *(const v8f*)(br + 64 + cb);
  v8f brn = *(const v8f*)(br + 128 + cb);
  // h^T B-fragment store slot (contiguous m-run of 8): kb=w>>1, g=w&1
  int hb_off = (((wave >> 1) * 32) + bn + 16 * (wave & 1)) * 16 + 8 * hi;
  // outfrag base pieces (attention A-frag: rows = t, k = concat col)
  int jg0 = coloff + cb;
  int ko0 = jg0 & 31, kbb = jg0 >> 5;
  int flo = (((ko0 >> 3) & 1) ? 16 : 0);
  int fmb = ((ko0 & 16) ? 8 : 0);

  float hreg[8];
  for (int i = 0; i < 8; ++i) hreg[i] = 0.0f;

  int t0 = forward ? 0 : (T_SZ - 1);
  v8h gzv = *(const v8h*)(gxrow + (long)t0 * G3H + cb);
  v8h grv = *(const v8h*)(gxrow + (long)t0 * G3H + 64 + cb);
  v8h gnv = *(const v8h*)(gxrow + (long)t0 * G3H + 128 + cb);
  int idc = idrow[t0];
  __syncthreads();

  for (int tt = 0; tt < T_SZ; ++tt) {
    int t = forward ? tt : (T_SZ - 1 - tt);
    int buf = tt & 1;
    // ---- recurrent matmul: gh^T tiles for this lane's gates (WMMA) ----
    v16h b0 = *(const v16h*)(&hB[buf][(0 * 32 + lane) * 16]);
    v16h b1 = *(const v16h*)(&hB[buf][(1 * 32 + lane) * 16]);
    v8f cz = {}, cr = {}, cn = {};
    cz = __builtin_amdgcn_wmma_f32_16x16x32_f16(false, afr[0][0], false, b0, (short)0, cz, false, false);
    cz = __builtin_amdgcn_wmma_f32_16x16x32_f16(false, afr[0][1], false, b1, (short)0, cz, false, false);
    cr = __builtin_amdgcn_wmma_f32_16x16x32_f16(false, afr[1][0], false, b0, (short)0, cr, false, false);
    cr = __builtin_amdgcn_wmma_f32_16x16x32_f16(false, afr[1][1], false, b1, (short)0, cr, false, false);
    cn = __builtin_amdgcn_wmma_f32_16x16x32_f16(false, afr[2][0], false, b0, (short)0, cn, false, false);
    cn = __builtin_amdgcn_wmma_f32_16x16x32_f16(false, afr[2][1], false, b1, (short)0, cn, false, false);
    // ---- prefetch gx 8 steps ahead (HBM -> cache) ----
    if (tt + 8 < T_SZ) {
      int tp = forward ? (tt + 8) : (T_SZ - 9 - tt);
      const _Float16* p = gxrow + (long)tp * G3H;
      __builtin_prefetch(p + cb, 0, 1);
      __builtin_prefetch(p + 64 + cb, 0, 1);
      __builtin_prefetch(p + 128 + cb, 0, 1);
    }
    // ---- issue next step's gx/id loads now (consumed next iteration) ----
    v8h nz = gzv, nr = grv, nn = gnv; int idn = idc;
    if (tt + 1 < T_SZ) {
      int t1 = forward ? (tt + 1) : (T_SZ - 2 - tt);
      const _Float16* p = gxrow + (long)t1 * G3H;
      nz = *(const v8h*)(p + cb);
      nr = *(const v8h*)(p + 64 + cb);
      nn = *(const v8h*)(p + 128 + cb);
      idn = idrow[t1];
    }
    // ---- gate math, fully in registers ----
    bool msk = idc != 0;
    v8f outv; v8h hstore;
    for (int i = 0; i < 8; ++i) {
      float zg = 1.0f / (1.0f + __expf(-((float)gzv[i] + cz[i] + brz[i])));
      float rg = 1.0f / (1.0f + __expf(-((float)grv[i] + cr[i] + brr[i])));
      float hh = tanhf((float)gnv[i] + rg * (cn[i] + brn[i]));
      float hnew = msk ? (zg * hreg[i] + (1.0f - zg) * hh) : hreg[i];
      hreg[i] = hnew;
      outv[i] = hnew;
      hstore[i] = (_Float16)hnew;
    }
    long tok = (long)b * T_SZ + t;
    *(v8f*)(out32 + tok * 128 + jg0) = outv;
    *(v8h*)(&hB[buf ^ 1][hb_off]) = hstore;
    int mt = t >> 4, tl = t & 15;
    long fb = (((long)(b * MT_ATT + mt) * 4 + kbb) * 32 + (tl + flo)) * 16 + fmb;
    *(v8h*)(outfrag + fb) = hstore;
    gzv = nz; grv = nr; gnv = nn; idc = idn;
    __syncthreads();
  }
  if (forward) {
    *(v8f*)(hout + (long)b * 64 + cb) = *(v8f*)hreg;
  }
}

// Attention: one block (128 thr / 4 waves) per batch element.
__global__ void __launch_bounds__(128) k_attn(const float* __restrict__ out32,
                       const _Float16* __restrict__ outfrag,
                       const _Float16* __restrict__ wkfrag,
                       const float* __restrict__ bk,
                       const float* __restrict__ hf,
                       const float* __restrict__ Wq,
                       const float* __restrict__ bq,
                       const float* __restrict__ We,
                       const float* __restrict__ be_p,
                       const int* __restrict__ ids,
                       float* __restrict__ ctx) {
  __shared__ float q[64];
  __shared__ float tbuf[16 * 64];
  __shared__ float energy[MT_ATT * 16];
  __shared__ float red[128];
  int tid = threadIdx.x, lane = tid & 31, wave = tid >> 5;
  int b = blockIdx.x;
  float be = be_p[0];
  if (tid < 64) {
    float acc = bq[tid];
    for (int k = 0; k < 64; ++k) acc += hf[(long)b * 64 + k] * Wq[k * 64 + tid];
    q[tid] = acc;
  }
  __syncthreads();
  // Preload Wk B-fragments: wave w handles N cols [16w,16w+16).
  v16h bf[4];
  for (int kb = 0; kb < 4; ++kb)
    bf[kb] = *(const v16h*)(wkfrag + (((wave * 4 + kb) * 32) + lane) * 16);
  for (int mt = 0; mt < MT_ATT; ++mt) {
    v8f c = {};
    for (int kb = 0; kb < 4; ++kb) {
      v16h a = *(const v16h*)(outfrag + (((long)(b * MT_ATT + mt) * 4 + kb) * 32 + lane) * 16);
      c = __builtin_amdgcn_wmma_f32_16x16x32_f16(false, a, false, bf[kb], (short)0, c, false, false);
    }
    int col = wave * 16 + (lane & 15);
    float w_e = We[col], b_k = bk[col], q_c = q[col];
    for (int r = 0; r < 8; ++r) {
      int tr = r + ((lane >= 16) ? 8 : 0);
      tbuf[tr * 64 + col] = tanhf(c[r] + b_k + q_c) * w_e;
    }
    __syncthreads();
    if (tid < 16) {
      int t = mt * 16 + tid;
      float e = -1.0e9f;
      if (t < T_SZ) {
        float s = be;
        for (int j2 = 0; j2 < 64; ++j2) s += tbuf[tid * 64 + j2];
        e = s + ((ids[(long)b * T_SZ + t] != 0) ? 0.0f : -1.0e9f);
      }
      energy[t] = e;
    }
    __syncthreads();
  }
  // softmax over T (pad rows already -1e9)
  float mx = -1.0e30f;
  for (int t = tid; t < MT_ATT * 16; t += 128) mx = fmaxf(mx, energy[t]);
  red[tid] = mx; __syncthreads();
  for (int s = 64; s > 0; s >>= 1) { if (tid < s) red[tid] = fmaxf(red[tid], red[tid + s]); __syncthreads(); }
  float gmax = red[0]; __syncthreads();
  float sm = 0.0f;
  for (int t = tid; t < MT_ATT * 16; t += 128) { float e = __expf(energy[t] - gmax); energy[t] = e; sm += e; }
  red[tid] = sm; __syncthreads();
  for (int s = 64; s > 0; s >>= 1) { if (tid < s) red[tid] += red[tid + s]; __syncthreads(); }
  float inv = 1.0f / red[0];
  __syncthreads();
  // context: thread j accumulates over t
  float acc = 0.0f;
  const float* orow = out32 + (long)b * T_SZ * 128 + tid;
  for (int t = 0; t < T_SZ; ++t) acc += energy[t] * orow[(long)t * 128];
  ctx[(long)b * 128 + tid] = acc * inv;
}

// ---------------------------------------------------------------------------

static inline size_t al256(size_t x) { return (x + 255) & ~(size_t)255; }

extern "C" void kernel_launch(void* const* d_in, const int* in_sizes, int n_in,
                              void* d_out, int out_size, void* d_ws, size_t ws_size,
                              hipStream_t stream) {
  const int*   ids  = (const int*)  d_in[0];
  const float* emb  = (const float*)d_in[1];
  const float* Wx_f = (const float*)d_in[2];
  const float* Wh_f = (const float*)d_in[3];
  const float* bi_f = (const float*)d_in[4];
  const float* br_f = (const float*)d_in[5];
  const float* Wx_b = (const float*)d_in[6];
  const float* Wh_b = (const float*)d_in[7];
  const float* bi_b = (const float*)d_in[8];
  const float* br_b = (const float*)d_in[9];
  const float* Wk   = (const float*)d_in[10];
  const float* bk   = (const float*)d_in[11];
  const float* Wq   = (const float*)d_in[12];
  const float* bq   = (const float*)d_in[13];
  const float* We   = (const float*)d_in[14];
  const float* be   = (const float*)d_in[15];
  float* out = (float*)d_out;

  char* ws = (char*)d_ws;
  size_t off = 0;
  auto take = [&](size_t bytes) { char* p = ws + off; off = al256(off + bytes); return p; };

  _Float16* xfrag   = (_Float16*)take((size_t)BT * 64 * 2);
  _Float16* wxf     = (_Float16*)take(12 * 2 * 512 * 2);
  _Float16* wxb     = (_Float16*)take(12 * 2 * 512 * 2);
  _Float16* whAf    = (_Float16*)take(12 * 2 * 512 * 2);
  _Float16* whAb    = (_Float16*)take(12 * 2 * 512 * 2);
  _Float16* wkf     = (_Float16*)take(4 * 4 * 512 * 2);
  _Float16* gx_f    = (_Float16*)take((size_t)BT * G3H * 2);
  _Float16* gx_b    = (_Float16*)take((size_t)BT * G3H * 2);
  float*    out32   = (float*)   take((size_t)BT * 128 * 4);
  _Float16* outfrag = (_Float16*)take((size_t)B_SZ * MT_ATT * 4 * 512 * 2);
  float*    h_f     = (float*)   take((size_t)B_SZ * 64 * 4);
  (void)ws_size; (void)in_sizes; (void)n_in; (void)out_size;

  // 1) embedding gather -> A-fragment f16
  k_gather<<<(BT + 255) / 256, 256, 0, stream>>>(ids, emb, xfrag);

  // 2) pack weights: Wx, Wk -> B-frag; Wh^T -> A-frag
  k_packB<<<(12 * 2 * 512 + 255) / 256, 256, 0, stream>>>(Wx_f, wxf, E_SZ, G3H, 64);
  k_packB<<<(12 * 2 * 512 + 255) / 256, 256, 0, stream>>>(Wx_b, wxb, E_SZ, G3H, 64);
  k_packA_T<<<(12 * 2 * 512 + 255) / 256, 256, 0, stream>>>(Wh_f, whAf);
  k_packA_T<<<(12 * 2 * 512 + 255) / 256, 256, 0, stream>>>(Wh_b, whAb);
  k_packB<<<(4 * 4 * 512 + 255) / 256, 256, 0, stream>>>(Wk, wkf, 128, 64, 128);

  // 3) gx GEMMs (WMMA)
  k_gemm_gx<<<BT / 128, 256, 0, stream>>>(xfrag, wxf, bi_f, gx_f);
  k_gemm_gx<<<BT / 128, 256, 0, stream>>>(xfrag, wxb, bi_b, gx_b);

  // 4) forward + backward GRU scans (transposed WMMA recurrence, pipelined gx)
  k_scan<<<B_SZ / 16, 128, 0, stream>>>(gx_f, whAf, br_f, ids, out32, outfrag, h_f, 1, 0);
  k_scan<<<B_SZ / 16, 128, 0, stream>>>(gx_b, whAb, br_b, ids, out32, outfrag, h_f, 0, 64);

  // 5) fused attention (WMMA keys + softmax + context)
  k_attn<<<B_SZ, 128, 0, stream>>>(out32, outfrag, wkf, bk, h_f, Wq, bq, We, be, ids, out);
}
